// SparseMoE_19928648254011
// MI455X (gfx1250) — matrix operand order repaired and checked by reading, not compiled, not run on hardware
//
#include <hip/hip_runtime.h>

#define Bsz   32
#define Cch   128
#define Hdim  56
#define Wdim  56
#define HW    (Hdim * Wdim)          // 3136
#define Eexp  8
#define HID   32
#define OUT_ELEMS (Bsz * Cch * HW)   // 12845056

typedef __attribute__((ext_vector_type(16))) __bf16          v16bf;
typedef __attribute__((ext_vector_type(16))) unsigned short  v16u;
typedef __attribute__((ext_vector_type(8)))  float           v8f;

union Frag { v16u u; v16bf b; };

__device__ __forceinline__ unsigned short f2bf(float f) {
    union { float f; unsigned u; } c; c.f = f;
    unsigned r = c.u + 0x7FFFu + ((c.u >> 16) & 1u);   // round-to-nearest-even
    return (unsigned short)(r >> 16);
}

// ---------------- 1) weight f32 -> bf16, pre-swizzled to WMMA A-fragment order --
// dst index: ((((((e*4 + cc)*3 + kh)*3 + kw)*128 + m)*2 + hl)*16 + el
// where k = el + 16*hl maps (hl,el) -> K in [0,32), cin = cc*32 + k.
__global__ void prep_swizzle_kernel(const float* __restrict__ w,
                                    unsigned short* __restrict__ o, int n) {
    int i = blockIdx.x * blockDim.x + threadIdx.x;
    if (i >= n) return;
    int el = i & 15;
    int hl = (i >> 4) & 1;
    int m  = (i >> 5) & 127;
    int t  = i >> 12;
    int kw = t % 3;  t /= 3;
    int kh = t % 3;  t /= 3;
    int cc = t & 3;
    int e  = t >> 2;
    int k   = el + 16 * hl;
    int cin = cc * 32 + k;
    float v = w[((((size_t)e * Cch + m) * Cch + cin) * 3 + kh) * 3 + kw];
    o[i] = f2bf(v);
}

// ---------------- 2) router: mean -> MLP -> softmax -> top2 ----------------
__global__ void router_kernel(const float* __restrict__ x,
                              const float* __restrict__ w1, const float* __restrict__ b1,
                              const float* __restrict__ w2, const float* __restrict__ b2,
                              float* __restrict__ probs,
                              int* __restrict__ tkidx, float* __restrict__ tkval) {
    __shared__ float sh[Cch];
    __shared__ float shid[HID];
    __shared__ float slog[Eexp];
    const int b = blockIdx.x;
    const int c = threadIdx.x;          // 128 threads

    const float* xc = x + ((size_t)(b * Cch + c)) * HW;
    float s = 0.f;
    for (int p = 0; p < HW; ++p) s += xc[p];
    sh[c] = s * (1.0f / (float)HW);
    __syncthreads();

    if (c < HID) {
        float a = b1[c];
        for (int i = 0; i < Cch; ++i) a += sh[i] * w1[c * Cch + i];
        shid[c] = fmaxf(a, 0.f);
    }
    __syncthreads();

    if (c < Eexp) {
        float a = b2[c];
        for (int i = 0; i < HID; ++i) a += shid[i] * w2[c * HID + i];
        slog[c] = a;
    }
    __syncthreads();

    if (c == 0) {
        float mx = slog[0];
        for (int e = 1; e < Eexp; ++e) mx = fmaxf(mx, slog[e]);
        float p[Eexp], sum = 0.f;
        for (int e = 0; e < Eexp; ++e) { p[e] = __expf(slog[e] - mx); sum += p[e]; }
        float inv = 1.0f / sum;
        for (int e = 0; e < Eexp; ++e) { p[e] *= inv; probs[b * Eexp + e] = p[e]; }
        int i0 = 0;
        for (int e = 1; e < Eexp; ++e) if (p[e] > p[i0]) i0 = e;
        int i1 = (i0 == 0) ? 1 : 0;
        for (int e = 0; e < Eexp; ++e) if (e != i0 && p[e] > p[i1]) i1 = e;
        tkidx[b * 2 + 0] = i0;  tkval[b * 2 + 0] = p[i0];
        tkidx[b * 2 + 1] = i1;  tkval[b * 2 + 1] = p[i1];
    }
}

// ---------------- 3) implicit-GEMM MoE conv via WMMA bf16 ----------------
// grid: 32 images x 49 (8x8) tiles.  block: 256 thr = 8 waves.
// wave w -> output channels [16w, 16w+16), 64 pixels, both experts.
__global__ __launch_bounds__(256) void moe_conv_kernel(
    const float* __restrict__ x, const unsigned short* __restrict__ wsw,
    const float* __restrict__ gamma, const float* __restrict__ beta,
    const float* __restrict__ rmean, const float* __restrict__ rvar,
    const int* __restrict__ tkidx, const float* __restrict__ tkval,
    float* __restrict__ out) {
    // pixel-major halo patch: patch[(y*10+x)*32 + cin]  (bf16) -> B frags contiguous
    __shared__ unsigned short patch[100 * 32];

    const int tid  = threadIdx.x;
    const int wave = tid >> 5;
    const int lane = tid & 31;
    const int hl   = lane >> 4;        // half-lane (wave32)
    const int ln   = lane & 15;

    const int b    = blockIdx.x / 49;
    const int tile = blockIdx.x % 49;
    const int h0   = (tile / 7) * 8;
    const int w0   = (tile % 7) * 8;

    const int   ex0 = tkidx[b * 2 + 0], ex1 = tkidx[b * 2 + 1];
    const float gv0 = tkval[b * 2 + 0], gv1 = tkval[b * 2 + 1];

    v8f acc[2][4];
    v8f zero = {0.f, 0.f, 0.f, 0.f, 0.f, 0.f, 0.f, 0.f};
#pragma unroll
    for (int j = 0; j < 2; ++j)
#pragma unroll
        for (int nt = 0; nt < 4; ++nt) acc[j][nt] = zero;

    const int m = wave * 16 + ln;      // this lane's A-row (output channel)

    for (int cc = 0; cc < 4; ++cc) {   // 4 chunks of 32 input channels
        if (cc < 3)
            __builtin_prefetch(&x[(((size_t)(b * Cch + (cc + 1) * 32)) * Hdim + h0) * Wdim + w0], 0, 0);
        __syncthreads();
        // stage 10x10 halo patch for 32 input channels, f32 -> bf16, pixel-major
        for (int t = tid; t < 3200; t += 256) {
            int ci = t / 100, r = t % 100, py = r / 10, px = r % 10;
            int gy = h0 + py - 1, gx = w0 + px - 1;
            float v = 0.f;
            if (gy >= 0 && gy < Hdim && gx >= 0 && gx < Wdim)
                v = x[(((size_t)(b * Cch + cc * 32 + ci)) * Hdim + gy) * Wdim + gx];
            patch[r * 32 + ci] = f2bf(v);
        }
        __syncthreads();

        for (int kh = 0; kh < 3; ++kh) {
            for (int kw = 0; kw < 3; ++kw) {
                // A fragments: one contiguous 32B load per lane per expert
                size_t abase0 = ((((((size_t)ex0 * 4 + cc) * 3 + kh) * 3 + kw) * 128 + m) * 2 + hl) * 16;
                size_t abase1 = ((((((size_t)ex1 * 4 + cc) * 3 + kh) * 3 + kw) * 128 + m) * 2 + hl) * 16;
                Frag A0, A1;
                A0.u = *(const v16u*)(wsw + abase0);
                A1.u = *(const v16u*)(wsw + abase1);
#pragma unroll
                for (int nt = 0; nt < 4; ++nt) {
                    // B fragment: contiguous 32B LDS read (k = el + 16*hl)
                    Frag Bf;
                    int pl = nt * 16 + ln, py = pl >> 3, px = pl & 7;
                    Bf.u = *(const v16u*)(patch + ((py + kh) * 10 + (px + kw)) * 32 + 16 * hl);
                    acc[0][nt] = __builtin_amdgcn_wmma_f32_16x16x32_bf16(
                        false, A0.b, false, Bf.b, (short)0, acc[0][nt], false, false);
                    acc[1][nt] = __builtin_amdgcn_wmma_f32_16x16x32_bf16(
                        false, A1.b, false, Bf.b, (short)0, acc[1][nt], false, false);
                }
            }
        }
    }

    // epilogue: BN affine + ReLU + gate + residual
#pragma unroll
    for (int nt = 0; nt < 4; ++nt) {
        int pl = nt * 16 + ln, py = pl >> 3, px = pl & 7;
        int gy = h0 + py, gx = w0 + px;
#pragma unroll
        for (int r = 0; r < 8; ++r) {
            int cout = wave * 16 + r + 8 * hl;      // C/D layout: M = r + 8*half
            size_t oi = (((size_t)(b * Cch + cout)) * Hdim + gy) * Wdim + gx;
            float res = x[oi];
            {
                int ix = ex0 * Cch + cout;
                float sc = gamma[ix] * rsqrtf(rvar[ix] + 1e-5f);
                float o  = (acc[0][nt][r] - rmean[ix]) * sc + beta[ix];
                res += fmaxf(o, 0.f) * gv0;
            }
            {
                int ix = ex1 * Cch + cout;
                float sc = gamma[ix] * rsqrtf(rvar[ix] + 1e-5f);
                float o  = (acc[1][nt][r] - rmean[ix]) * sc + beta[ix];
                res += fmaxf(o, 0.f) * gv1;
            }
            out[oi] = res;
        }
    }
}

extern "C" void kernel_launch(void* const* d_in, const int* in_sizes, int n_in,
                              void* d_out, int out_size, void* d_ws, size_t ws_size,
                              hipStream_t stream) {
    const float* x     = (const float*)d_in[0];
    const float* w1    = (const float*)d_in[1];
    const float* b1    = (const float*)d_in[2];
    const float* w2    = (const float*)d_in[3];
    const float* b2    = (const float*)d_in[4];
    const float* wexp  = (const float*)d_in[5];
    const float* gamma = (const float*)d_in[6];
    const float* beta  = (const float*)d_in[7];
    const float* rmean = (const float*)d_in[8];
    const float* rvar  = (const float*)d_in[9];

    float* out   = (float*)d_out;
    float* probs = out + (size_t)OUT_ELEMS;

    const int nW = Eexp * Cch * Cch * 9;               // 1,179,648 weights
    unsigned short* wsw = (unsigned short*)d_ws;       // 2,359,296 B, fragment-order
    char* wsb   = (char*)d_ws;
    int*   tkidx = (int*)(wsb + (size_t)nW * 2);
    float* tkval = (float*)(wsb + (size_t)nW * 2 + 256);

    prep_swizzle_kernel<<<(nW + 255) / 256, 256, 0, stream>>>(wexp, wsw, nW);
    router_kernel<<<Bsz, Cch, 0, stream>>>(x, w1, b1, w2, b2, probs, tkidx, tkval);
    moe_conv_kernel<<<Bsz * 49, 256, 0, stream>>>(x, wsw, gamma, beta, rmean, rvar,
                                                  tkidx, tkval, out);
}
